// HybridAttention_15418932593312
// MI455X (gfx1250) — matrix operand order repaired and checked by reading, not compiled, not run on hardware
//
#include <hip/hip_runtime.h>
#include <hip/hip_bf16.h>

// ---------------------------------------------------------------------------
// CDNA5 (gfx1250) hybrid local+global attention, bf16 WMMA f32-accumulate.
// ---------------------------------------------------------------------------

typedef __bf16 bf16_t;
typedef __attribute__((ext_vector_type(16))) __bf16 v16bf;
typedef __attribute__((ext_vector_type(8)))  float  v8f;

constexpr int Bc    = 2;
constexpr int Sc    = 2048;
constexpr int Dc    = 1024;
constexpr int Hc    = 16;
constexpr int DHc   = 64;
constexpr int WINc  = 64;
constexpr int CHc   = Sc / WINc;     // 32 chunks
constexpr int MTOT  = Bc * Sc;       // 4096 rows
constexpr int QKVN  = 3 * Dc;        // 3072
constexpr float SCALE  = 0.125f;     // 1/sqrt(64)
constexpr float NEGBIG = -1e9f;

// ---- workspace layout (bytes) ---------------------------------------------
constexpr size_t SZ_XLN  = (size_t)MTOT * Dc * 2;        // 8 MB
constexpr size_t SZ_WQKV = (size_t)QKVN * Dc * 2;        // 6 MB
constexpr size_t SZ_WO   = (size_t)Dc * Dc * 2;          // 2 MB
constexpr size_t SZ_QKV  = (size_t)MTOT * QKVN * 2;      // 24 MB
constexpr size_t SZ_O    = (size_t)MTOT * Dc * 2;        // 8 MB

constexpr size_t OFF_XLN_L  = 0;
constexpr size_t OFF_XLN_G  = OFF_XLN_L  + SZ_XLN;
constexpr size_t OFF_WQKV_L = OFF_XLN_G  + SZ_XLN;
constexpr size_t OFF_WQKV_G = OFF_WQKV_L + SZ_WQKV;
constexpr size_t OFF_WO_L   = OFF_WQKV_G + SZ_WQKV;
constexpr size_t OFF_WO_G   = OFF_WO_L   + SZ_WO;
constexpr size_t OFF_QKV_L  = OFF_WO_G   + SZ_WO;
constexpr size_t OFF_QKV_G  = OFF_QKV_L  + SZ_QKV;
constexpr size_t OFF_O_L    = OFF_QKV_G  + SZ_QKV;
constexpr size_t OFF_O_G    = OFF_O_L    + SZ_O;

// ---- WMMA helpers ----------------------------------------------------------
__device__ inline v8f wmma_bf16(v16bf a, v16bf b, v8f c) {
  // D = A(16x32 bf16) * B(32x16 bf16) + C(16x16 f32)
  return __builtin_amdgcn_wmma_f32_16x16x32_bf16(
      /*neg_a=*/false, a, /*neg_b=*/false, b,
      /*c_mod=*/(short)0, c, /*reuse_a=*/false, /*reuse_b=*/false);
}

__device__ inline v8f vzero8() {
  v8f z = {0.f, 0.f, 0.f, 0.f, 0.f, 0.f, 0.f, 0.f};
  return z;
}

union Frag16 { v16bf v; struct { int4 lo; int4 hi; } q; };

// A-frag 16x32 (MxK) bf16: lane holds row M=lane&15; lanes 0-15 cover
// K = {0..7,16..23}, lanes 16-31 cover K = {8..15,24..31}.
// tile: row-major [16][>=32] at origin, ld = row stride (elements).
__device__ inline v16bf load_a16x32(const bf16_t* tile, int ld) {
  const int lane = threadIdx.x & 31;
  const int row = lane & 15;
  const int hi  = lane >> 4;
  const bf16_t* p = tile + (size_t)row * ld + hi * 8;
  Frag16 u;
  u.q.lo = *reinterpret_cast<const int4*>(p);       // K = hi*8 + 0..7
  u.q.hi = *reinterpret_cast<const int4*>(p + 16);  // K = hi*8 + 16..23
  return u.v;
}

// B-frag 32x16 (KxN) bf16: lane holds column N=lane&15; lanes 0-15 cover
// K = {0..7,16..23}, lanes 16-31 cover K = {8..15,24..31}. Source is N-major with K contiguous:
// element (k,n) = tile[n*ld + k]  (i.e. W[N][K] row-major, or K^T layouts).
__device__ inline v16bf load_b32x16(const bf16_t* tile, int ld) {
  const int lane = threadIdx.x & 31;
  const int n  = lane & 15;
  const int hi = lane >> 4;
  const bf16_t* p = tile + (size_t)n * ld + hi * 8;
  Frag16 u;
  u.q.lo = *reinterpret_cast<const int4*>(p);
  u.q.hi = *reinterpret_cast<const int4*>(p + 16);
  return u.v;
}

// ---------------------------------------------------------------------------
// Kernel 1: fp32 -> bf16 conversion (weights)
// ---------------------------------------------------------------------------
__global__ void k_cvt_bf16(const float* __restrict__ s, bf16_t* __restrict__ d, int n) {
  int i = blockIdx.x * blockDim.x + threadIdx.x;
  if (i < n) d[i] = (bf16_t)s[i];
}

// ---------------------------------------------------------------------------
// Kernel 2: fused dual LayerNorm (one stat pass, two gamma/beta outputs)
// one block (256 threads) per row of 1024
// ---------------------------------------------------------------------------
__global__ __launch_bounds__(256) void k_layernorm2(
    const float* __restrict__ x,
    const float* __restrict__ gl, const float* __restrict__ bl,
    const float* __restrict__ gg, const float* __restrict__ bg,
    bf16_t* __restrict__ xl, bf16_t* __restrict__ xg) {
  const int row = blockIdx.x;
  const float* xr = x + (size_t)row * Dc;
  float vals[4];
  float s = 0.f, s2 = 0.f;
#pragma unroll
  for (int i = 0; i < 4; ++i) {
    float v = xr[threadIdx.x + 256 * i];
    vals[i] = v;
    s += v; s2 += v * v;
  }
#pragma unroll
  for (int m = 16; m; m >>= 1) {
    s  += __shfl_xor(s, m);
    s2 += __shfl_xor(s2, m);
  }
  __shared__ float rs[8], rs2[8];
  const int w = threadIdx.x >> 5;
  if ((threadIdx.x & 31) == 0) { rs[w] = s; rs2[w] = s2; }
  __syncthreads();
  s = 0.f; s2 = 0.f;
#pragma unroll
  for (int i = 0; i < 8; ++i) { s += rs[i]; s2 += rs2[i]; }
  const float mean = s * (1.f / Dc);
  const float var  = s2 * (1.f / Dc) - mean * mean;
  const float rinv = rsqrtf(var + 1e-5f);
#pragma unroll
  for (int i = 0; i < 4; ++i) {
    int c = threadIdx.x + 256 * i;
    float nv = (vals[i] - mean) * rinv;
    xl[(size_t)row * Dc + c] = (bf16_t)(nv * gl[c] + bl[c]);
    xg[(size_t)row * Dc + c] = (bf16_t)(nv * gg[c] + bg[c]);
  }
}

// ---------------------------------------------------------------------------
// Kernel 3: GEMM  C(bf16)[M,N] = A(bf16)[M,K] * W(bf16)[N,K]^T + bias(f32)[N]
// block = 256 threads = 8 waves arranged 2(M) x 4(N); block tile 64x256,
// wave tile 32x64 (2x4 WMMA accumulators => 8 WMMA per 12 b128 loads/K-step).
// ---------------------------------------------------------------------------
__global__ __launch_bounds__(256) void k_gemm_bf16(
    const bf16_t* __restrict__ A, const bf16_t* __restrict__ W,
    const float* __restrict__ bias, bf16_t* __restrict__ C,
    int N, int K) {
  const int wave = threadIdx.x >> 5;
  const int wm = wave & 1, wn = wave >> 1;          // 2 x 4
  const int m0 = blockIdx.y * 64 + wm * 32;
  const int n0 = blockIdx.x * 256 + wn * 64;
  v8f acc[2][4];
#pragma unroll
  for (int i = 0; i < 2; ++i)
#pragma unroll
    for (int j = 0; j < 4; ++j) acc[i][j] = vzero8();

  for (int k0 = 0; k0 < K; k0 += 32) {
    if (k0 + 64 < K) {  // pull next-next K-tiles toward WGP$/GL2
      __builtin_prefetch(A + (size_t)m0 * K + k0 + 64, 0, 3);
      __builtin_prefetch(W + (size_t)n0 * K + k0 + 64, 0, 3);
    }
    v16bf fa[2], fb[4];
#pragma unroll
    for (int i = 0; i < 2; ++i)
      fa[i] = load_a16x32(A + (size_t)(m0 + 16 * i) * K + k0, K);
#pragma unroll
    for (int j = 0; j < 4; ++j)
      fb[j] = load_b32x16(W + (size_t)(n0 + 16 * j) * K + k0, K);
#pragma unroll
    for (int i = 0; i < 2; ++i)
#pragma unroll
      for (int j = 0; j < 4; ++j) acc[i][j] = wmma_bf16(fa[i], fb[j], acc[i][j]);
  }

  const int lane = threadIdx.x & 31;
  const int nl = lane & 15, mo = (lane >> 4) * 8;
#pragma unroll
  for (int j = 0; j < 4; ++j) {
    const float bv = bias[n0 + 16 * j + nl];
#pragma unroll
    for (int i = 0; i < 2; ++i)
#pragma unroll
      for (int r = 0; r < 8; ++r)
        C[(size_t)(m0 + 16 * i + mo + r) * N + n0 + 16 * j + nl] =
            (bf16_t)(acc[i][j][r] + bv);
  }
}

// ---------------------------------------------------------------------------
// Kernel 4: global attention, flash-style streaming softmax.
// grid = (S/128, H, B); block = 256 = 8 waves; wave owns a 16-row q tile.
// Per 32-key step: block stages V^T[64][32] in LDS (shared by all waves),
// 4 WMMA for scores, online max/sum rescale, P through per-wave LDS
// transpose buffer, 4 WMMA for P*V.
// ---------------------------------------------------------------------------
__global__ __launch_bounds__(256) void k_attn_global(
    const bf16_t* __restrict__ qkv, bf16_t* __restrict__ O) {
  const int h = blockIdx.y, b = blockIdx.z;
  const int w = threadIdx.x >> 5;
  const int lane = threadIdx.x & 31;
  const int nl = lane & 15, mo = (lane >> 4) * 8;

  __shared__ bf16_t vt[DHc][32];        // V^T for current 32-key chunk (4 KB)
  __shared__ bf16_t pls[8][16][32];     // per-wave P transpose buffer (8 KB)

  const size_t rs = QKVN;
  const bf16_t* Qb = qkv + (size_t)(b * Sc) * rs + (size_t)h * DHc;
  const bf16_t* Kb = Qb + Dc;
  const bf16_t* Vb = Qb + 2 * Dc;
  const int q0 = blockIdx.x * 128 + w * 16;

  const v16bf qa0 = load_a16x32(Qb + (size_t)q0 * rs, rs);
  const v16bf qa1 = load_a16x32(Qb + (size_t)q0 * rs + 32, rs);

  float m[8], l[8];
  v8f o[4];
#pragma unroll
  for (int r = 0; r < 8; ++r) { m[r] = -3e38f; l[r] = 0.f; }
#pragma unroll
  for (int t = 0; t < 4; ++t) o[t] = vzero8();

  for (int kt0 = 0; kt0 < Sc; kt0 += 32) {
    __syncthreads();
    // cooperative V^T stage: 32 keys x 64 dh = 2048 elems / 256 threads
#pragma unroll
    for (int e = 0; e < 8; ++e) {
      int idx = threadIdx.x + 256 * e;
      int kk = idx >> 6, n = idx & 63;
      vt[n][kk] = Vb[(size_t)(kt0 + kk) * rs + n];
    }
    __syncthreads();

    // scores: two 16x16 tiles (keys kt0..+15, kt0+16..+31)
    const bf16_t* kb0 = Kb + (size_t)kt0 * rs;
    const bf16_t* kb1 = Kb + (size_t)(kt0 + 16) * rs;
    v8f s0 = vzero8(), s1 = vzero8();
    s0 = wmma_bf16(qa0, load_b32x16(kb0, rs), s0);
    s0 = wmma_bf16(qa1, load_b32x16(kb0 + 32, rs), s0);
    s1 = wmma_bf16(qa0, load_b32x16(kb1, rs), s1);
    s1 = wmma_bf16(qa1, load_b32x16(kb1 + 32, rs), s1);
#pragma unroll
    for (int r = 0; r < 8; ++r) { s0[r] *= SCALE; s1[r] *= SCALE; }

    // online softmax: tile row-max -> running max, rescale
    float mn[8], f[8], ps[8];
#pragma unroll
    for (int r = 0; r < 8; ++r) mn[r] = fmaxf(s0[r], s1[r]);
#pragma unroll
    for (int msk = 1; msk < 16; msk <<= 1)
#pragma unroll
      for (int r = 0; r < 8; ++r) mn[r] = fmaxf(mn[r], __shfl_xor(mn[r], msk));
#pragma unroll
    for (int r = 0; r < 8; ++r) {
      float mnew = fmaxf(m[r], mn[r]);
      f[r] = __expf(m[r] - mnew);
      m[r] = mnew;
    }
#pragma unroll
    for (int r = 0; r < 8; ++r) {
      float p0 = __expf(s0[r] - m[r]);
      float p1 = __expf(s1[r] - m[r]);
      pls[w][mo + r][nl]      = (bf16_t)p0;   // same-wave LDS: in-order
      pls[w][mo + r][16 + nl] = (bf16_t)p1;
      ps[r] = p0 + p1;
    }
#pragma unroll
    for (int msk = 1; msk < 16; msk <<= 1)
#pragma unroll
      for (int r = 0; r < 8; ++r) ps[r] += __shfl_xor(ps[r], msk);
#pragma unroll
    for (int r = 0; r < 8; ++r) l[r] = l[r] * f[r] + ps[r];
#pragma unroll
    for (int t = 0; t < 4; ++t)
#pragma unroll
      for (int r = 0; r < 8; ++r) o[t][r] *= f[r];

    // P(16x32) * V(32x64): 4 WMMA (one per 16-wide dh slice)
    const v16bf pa = load_a16x32(&pls[w][0][0], 32);
#pragma unroll
    for (int nt = 0; nt < 4; ++nt)
      o[nt] = wmma_bf16(pa, load_b32x16(&vt[nt * 16][0], 32), o[nt]);
  }

#pragma unroll
  for (int nt = 0; nt < 4; ++nt)
#pragma unroll
    for (int r = 0; r < 8; ++r)
      O[(size_t)(b * Sc + q0 + mo + r) * Dc + h * DHc + nt * 16 + nl] =
          (bf16_t)(o[nt][r] / l[r]);
}

// ---------------------------------------------------------------------------
// Kernel 5: local (sliding-window) attention.
// grid = (C, H, B); block = 128 = 4 waves; wave owns 16 q rows of chunk c.
// Keys = 128 (chunk c-1 ++ chunk c); chunk 0 masks its first 64 keys.
// ---------------------------------------------------------------------------
__global__ __launch_bounds__(128) void k_attn_local(
    const bf16_t* __restrict__ qkv, bf16_t* __restrict__ O) {
  const int c = blockIdx.x, h = blockIdx.y, b = blockIdx.z;
  const int w = threadIdx.x >> 5;
  const int lane = threadIdx.x & 31;
  const int nl = lane & 15, mo = (lane >> 4) * 8;

  __shared__ bf16_t vt[DHc][2 * WINc];     // V^T [64][128]  (16 KB)
  __shared__ bf16_t pls[4][16][2 * WINc];  // per-wave P [16][128] (16 KB)

  const size_t rs = QKVN;
  const bf16_t* Qb = qkv + (size_t)(b * Sc) * rs + (size_t)h * DHc;
  const bf16_t* Kb = Qb + Dc;
  const bf16_t* Vb = Qb + 2 * Dc;

  const int q0 = c * WINc + w * 16;
  const int s0 = c * WINc - WINc;          // first key row (negative for c==0)

  // cooperative V^T stage: 128 keys x 64 dh = 8192 elems / 128 threads
#pragma unroll
  for (int e = 0; e < 64; ++e) {
    int idx = threadIdx.x + 128 * e;
    int kk = idx >> 6, n = idx & 63;       // kk 0..127, n 0..63
    int srow = s0 + kk; if (srow < 0) srow = 0;  // c==0 prefix gets p=0 anyway
    vt[n][kk] = Vb[(size_t)srow * rs + n];
  }
  __syncthreads();

  const v16bf qa0 = load_a16x32(Qb + (size_t)q0 * rs, rs);
  const v16bf qa1 = load_a16x32(Qb + (size_t)q0 * rs + 32, rs);

  // scores: 8 tiles of 16 keys
  v8f sc[8];
  const int ktMin = (c == 0) ? 4 : 0;
#pragma unroll
  for (int kt = 0; kt < 8; ++kt) {
    if (kt < ktMin) {
#pragma unroll
      for (int r = 0; r < 8; ++r) sc[kt][r] = NEGBIG;
    } else {
      const bf16_t* kb = Kb + (size_t)(s0 + kt * 16) * rs;
      v8f s = vzero8();
      s = wmma_bf16(qa0, load_b32x16(kb, rs), s);
      s = wmma_bf16(qa1, load_b32x16(kb + 32, rs), s);
#pragma unroll
      for (int r = 0; r < 8; ++r) sc[kt][r] = s[r] * SCALE;
    }
  }

  // softmax over 128 keys
  float m[8], l[8];
#pragma unroll
  for (int r = 0; r < 8; ++r) m[r] = NEGBIG;
#pragma unroll
  for (int kt = 0; kt < 8; ++kt)
#pragma unroll
    for (int r = 0; r < 8; ++r) m[r] = fmaxf(m[r], sc[kt][r]);
#pragma unroll
  for (int msk = 1; msk < 16; msk <<= 1)
#pragma unroll
    for (int r = 0; r < 8; ++r) m[r] = fmaxf(m[r], __shfl_xor(m[r], msk));
#pragma unroll
  for (int r = 0; r < 8; ++r) l[r] = 0.f;
#pragma unroll
  for (int kt = 0; kt < 8; ++kt)
#pragma unroll
    for (int r = 0; r < 8; ++r) {
      float p = __expf(sc[kt][r] - m[r]);
      l[r] += p;
      pls[w][mo + r][kt * 16 + nl] = (bf16_t)p;  // same-wave LDS: in-order
    }
#pragma unroll
  for (int msk = 1; msk < 16; msk <<= 1)
#pragma unroll
    for (int r = 0; r < 8; ++r) l[r] += __shfl_xor(l[r], msk);

  // O(16x64) = P(16x128) * V(128x64)
  v8f o[4];
#pragma unroll
  for (int t = 0; t < 4; ++t) o[t] = vzero8();
#pragma unroll
  for (int kc = 0; kc < 4; ++kc) {
    v16bf pa = load_a16x32(&pls[w][0][kc * 32], 2 * WINc);
#pragma unroll
    for (int nt = 0; nt < 4; ++nt)
      o[nt] = wmma_bf16(pa, load_b32x16(&vt[nt * 16][kc * 32], 2 * WINc), o[nt]);
  }
#pragma unroll
  for (int nt = 0; nt < 4; ++nt)
#pragma unroll
    for (int r = 0; r < 8; ++r)
      O[(size_t)(b * Sc + q0 + mo + r) * Dc + h * DHc + nt * 16 + nl] =
          (bf16_t)(o[nt][r] / l[r]);
}

// ---------------------------------------------------------------------------
// Kernel 6: fused output projection.
// out(f32)[M,D] = Ol*Wol^T + Og*Wog^T + bol + bog   (two K-loops, one tile)
// Same 64x256 block / 32x64 wave tiling as k_gemm_bf16.
// ---------------------------------------------------------------------------
__global__ __launch_bounds__(256) void k_gemm_out(
    const bf16_t* __restrict__ Al, const bf16_t* __restrict__ Wl,
    const float* __restrict__ bl,
    const bf16_t* __restrict__ Ag, const bf16_t* __restrict__ Wg,
    const float* __restrict__ bg, float* __restrict__ Out) {
  constexpr int N = Dc, K = Dc;
  const int wave = threadIdx.x >> 5;
  const int wm = wave & 1, wn = wave >> 1;
  const int m0 = blockIdx.y * 64 + wm * 32;
  const int n0 = blockIdx.x * 256 + wn * 64;
  v8f acc[2][4];
#pragma unroll
  for (int i = 0; i < 2; ++i)
#pragma unroll
    for (int j = 0; j < 4; ++j) acc[i][j] = vzero8();

  for (int k0 = 0; k0 < K; k0 += 32) {
    v16bf fa[2], fb[4];
#pragma unroll
    for (int i = 0; i < 2; ++i)
      fa[i] = load_a16x32(Al + (size_t)(m0 + 16 * i) * K + k0, K);
#pragma unroll
    for (int j = 0; j < 4; ++j)
      fb[j] = load_b32x16(Wl + (size_t)(n0 + 16 * j) * K + k0, K);
#pragma unroll
    for (int i = 0; i < 2; ++i)
#pragma unroll
      for (int j = 0; j < 4; ++j) acc[i][j] = wmma_bf16(fa[i], fb[j], acc[i][j]);
  }
  for (int k0 = 0; k0 < K; k0 += 32) {
    v16bf fa[2], fb[4];
#pragma unroll
    for (int i = 0; i < 2; ++i)
      fa[i] = load_a16x32(Ag + (size_t)(m0 + 16 * i) * K + k0, K);
#pragma unroll
    for (int j = 0; j < 4; ++j)
      fb[j] = load_b32x16(Wg + (size_t)(n0 + 16 * j) * K + k0, K);
#pragma unroll
    for (int i = 0; i < 2; ++i)
#pragma unroll
      for (int j = 0; j < 4; ++j) acc[i][j] = wmma_bf16(fa[i], fb[j], acc[i][j]);
  }

  const int lane = threadIdx.x & 31;
  const int nl = lane & 15, mo = (lane >> 4) * 8;
#pragma unroll
  for (int j = 0; j < 4; ++j) {
    const float bv = bl[n0 + 16 * j + nl] + bg[n0 + 16 * j + nl];
#pragma unroll
    for (int i = 0; i < 2; ++i)
#pragma unroll
      for (int r = 0; r < 8; ++r)
        Out[(size_t)(m0 + 16 * i + mo + r) * N + n0 + 16 * j + nl] =
            acc[i][j][r] + bv;
  }
}

// ---------------------------------------------------------------------------
// Host-side orchestration
// ---------------------------------------------------------------------------
extern "C" void kernel_launch(void* const* d_in, const int* in_sizes, int n_in,
                              void* d_out, int out_size, void* d_ws, size_t ws_size,
                              hipStream_t stream) {
  (void)in_sizes; (void)n_in; (void)out_size; (void)ws_size;
  const float* x      = (const float*)d_in[0];
  const float* ln_l_g = (const float*)d_in[1];
  const float* ln_l_b = (const float*)d_in[2];
  const float* Wqkv_l = (const float*)d_in[3];
  const float* bqkv_l = (const float*)d_in[4];
  const float* Wo_l   = (const float*)d_in[5];
  const float* bo_l   = (const float*)d_in[6];
  const float* ln_g_g = (const float*)d_in[7];
  const float* ln_g_b = (const float*)d_in[8];
  const float* Wqkv_g = (const float*)d_in[9];
  const float* bqkv_g = (const float*)d_in[10];
  const float* Wo_g   = (const float*)d_in[11];
  const float* bo_g   = (const float*)d_in[12];
  float* out = (float*)d_out;

  char* ws = (char*)d_ws;
  bf16_t* xln_l  = (bf16_t*)(ws + OFF_XLN_L);
  bf16_t* xln_g  = (bf16_t*)(ws + OFF_XLN_G);
  bf16_t* wqkvlb = (bf16_t*)(ws + OFF_WQKV_L);
  bf16_t* wqkvgb = (bf16_t*)(ws + OFF_WQKV_G);
  bf16_t* wolb   = (bf16_t*)(ws + OFF_WO_L);
  bf16_t* wogb   = (bf16_t*)(ws + OFF_WO_G);
  bf16_t* qkv_l  = (bf16_t*)(ws + OFF_QKV_L);
  bf16_t* qkv_g  = (bf16_t*)(ws + OFF_QKV_G);
  bf16_t* o_l    = (bf16_t*)(ws + OFF_O_L);
  bf16_t* o_g    = (bf16_t*)(ws + OFF_O_G);

  // 1) weight fp32 -> bf16
  {
    int nq = QKVN * Dc, no = Dc * Dc;
    k_cvt_bf16<<<(nq + 255) / 256, 256, 0, stream>>>(Wqkv_l, wqkvlb, nq);
    k_cvt_bf16<<<(nq + 255) / 256, 256, 0, stream>>>(Wqkv_g, wqkvgb, nq);
    k_cvt_bf16<<<(no + 255) / 256, 256, 0, stream>>>(Wo_l, wolb, no);
    k_cvt_bf16<<<(no + 255) / 256, 256, 0, stream>>>(Wo_g, wogb, no);
  }

  // 2) fused dual layernorm
  k_layernorm2<<<MTOT, 256, 0, stream>>>(x, ln_l_g, ln_l_b, ln_g_g, ln_g_b,
                                         xln_l, xln_g);

  // 3) QKV projections (M=4096, N=3072, K=1024), block tile 64x256
  {
    dim3 grid(QKVN / 256, MTOT / 64);
    k_gemm_bf16<<<grid, 256, 0, stream>>>(xln_l, wqkvlb, bqkv_l, qkv_l, QKVN, Dc);
    k_gemm_bf16<<<grid, 256, 0, stream>>>(xln_g, wqkvgb, bqkv_g, qkv_g, QKVN, Dc);
  }

  // 4) attention
  k_attn_local<<<dim3(CHc, Hc, Bc), 128, 0, stream>>>(qkv_l, o_l);
  k_attn_global<<<dim3(Sc / 128, Hc, Bc), 256, 0, stream>>>(qkv_g, o_g);

  // 5) fused output projection + add
  {
    dim3 grid(Dc / 256, MTOT / 64);
    k_gemm_out<<<grid, 256, 0, stream>>>(o_l, wolb, bo_l, o_g, wogb, bo_g, out);
  }
}
